// GraphAttention2_71760313582233
// MI455X (gfx1250) — compile-verified, matrix operand order checked by reading, not run on hardware
//
#include <hip/hip_runtime.h>
#include <hip/hip_bf16.h>

// GAT layer for MI455X (gfx1250), flash-softmax fused with bf16 WMMA.
// N=6144 nodes, F=128 in-features, C=64 channels, H=4 heads.

#define Nn 6144
#define Ff 128
#define Cc 64
#define Hh 4
#define NEGC (-1.0e10f)

typedef __attribute__((ext_vector_type(16))) __bf16 v16bf;
typedef __attribute__((ext_vector_type(8)))  float v8f;

union BFrag {
    unsigned short us[16];
    uint4          q[2];
    v16bf          v;
};

// f32 -> bf16 bits, round-to-nearest-even
static __device__ __forceinline__ unsigned short bf16b(float f) {
    unsigned u = __builtin_bit_cast(unsigned, f);
    u += 0x7FFFu + ((u >> 16) & 1u);
    return (unsigned short)(u >> 16);
}

// ISA 7.12.2: 16-bit A/B fragment k index for element e (0..15) of a lane.
// v = e>>1 ; k = (v&3)*2 + (v>>2)*16 + hlf*8 + (e&1)   (hlf = lane>=16)
// => elements e=0..7 cover k = 8*hlf + e, e=8..15 cover k = 16 + 8*hlf + (e-8),
//    i.e. two contiguous 8-element runs -> two 16B vector loads per fragment.
static __device__ __forceinline__ int frag_k(int e, int hlf) {
    int v = e >> 1;
    return (v & 3) * 2 + (v >> 2) * 16 + hlf * 8 + (e & 1);
}

// ---------------------------------------------------------------------------
// Kernel 0: one-shot re-layout of W1/W2 into bf16 WMMA B-fragments.
// Wf linear index: ((G*2 + m)*32 + lane)*16 + e  with G = (h*4 + s)*4 + t
//   (h=head, s=k-slice of 32, t=16-col tile, m=0:W1 / 1:W2)
// ---------------------------------------------------------------------------
__global__ __launch_bounds__(256) void gat_prep_w(
    const float* __restrict__ W1, const float* __restrict__ W2,
    unsigned short* __restrict__ Wf)
{
    int idx  = blockIdx.x * 256 + threadIdx.x;      // 0 .. 65535
    int e    = idx & 15;
    int lane = (idx >> 4) & 31;
    int mm   = (idx >> 9) & 1;
    int t    = (idx >> 10) & 3;
    int s    = (idx >> 12) & 3;
    int h    = (idx >> 14) & 3;
    int hlf  = lane >> 4, l16 = lane & 15;
    int k    = s * 32 + frag_k(e, hlf);
    const float* W = mm ? W2 : W1;
    Wf[idx] = bf16b(W[(h * Ff + k) * Cc + t * 16 + l16]);
}

// ---------------------------------------------------------------------------
// Kernel 1: per-head projections via WMMA.
//   featsT[h][c][n]  (bf16)   : X@W1[h] transposed, B-operand for kernel 2
//   att_s[h][n], att_n[h][n]  : feats . a_self / a_neigh
//   S2[n][c]                  : sum_h (X@W2[h] + bias[h])
// Block = 128 threads (4 waves); wave h computes head h for a 16-row tile.
// ---------------------------------------------------------------------------
__global__ __launch_bounds__(128) void gat_proj(
    const float* __restrict__ X, const unsigned short* __restrict__ Wf,
    const float* __restrict__ a_self, const float* __restrict__ a_neigh,
    const float* __restrict__ bias,
    unsigned short* __restrict__ featsT, float* __restrict__ att_s,
    float* __restrict__ att_n, float* __restrict__ S2)
{
    __shared__ __align__(16) unsigned short xt[16 * Ff];   // X tile as bf16
    __shared__ float fts[Hh][16][Cc];                      // f32 feats per head
    __shared__ float s2[16][Cc];

    const int tid  = threadIdx.x;
    const int h    = tid >> 5;                  // wave == head
    const int lane = tid & 31;
    const int hlf  = lane >> 4;
    const int l16  = lane & 15;
    const int kb0  = 8 * hlf;
    const int i0   = blockIdx.x * 16;

    for (int idx = tid; idx < 16 * Ff; idx += 128) {
        int r = idx >> 7, f = idx & 127;
        xt[idx] = bf16b(X[(i0 + r) * Ff + f]);
    }
    for (int idx = tid; idx < 16 * Cc; idx += 128)
        ((float*)s2)[idx] = 0.0f;
    __syncthreads();

    v8f acc1[4], acc2[4];
    #pragma unroll
    for (int t = 0; t < 4; ++t)
        #pragma unroll
        for (int v = 0; v < 8; ++v) { acc1[t][v] = 0.0f; acc2[t][v] = 0.0f; }

    #pragma unroll
    for (int s = 0; s < 4; ++s) {               // k-slices of 32
        BFrag af;
        const unsigned short* xrow = xt + l16 * Ff + s * 32;
        af.q[0] = *(const uint4*)(xrow + kb0);
        af.q[1] = *(const uint4*)(xrow + 16 + kb0);

        #pragma unroll
        for (int t = 0; t < 4; ++t) {
            int G = (h * 4 + s) * 4 + t;
            const unsigned short* wb = Wf + G * 1024 + lane * 16;
            BFrag b1, b2;
            b1.q[0] = ((const uint4*)wb)[0];
            b1.q[1] = ((const uint4*)wb)[1];
            b2.q[0] = ((const uint4*)(wb + 512))[0];
            b2.q[1] = ((const uint4*)(wb + 512))[1];
            acc1[t] = __builtin_amdgcn_wmma_f32_16x16x32_bf16(
                false, af.v, false, b1.v, (short)0, acc1[t], false, false);
            acc2[t] = __builtin_amdgcn_wmma_f32_16x16x32_bf16(
                false, af.v, false, b2.v, (short)0, acc2[t], false, false);
        }
    }

    // spill feats to LDS; fold feats2+bias into shared S2 accumulator
    #pragma unroll
    for (int t = 0; t < 4; ++t)
        #pragma unroll
        for (int v = 0; v < 8; ++v) {
            int r = v + 8 * hlf, c = t * 16 + l16;
            fts[h][r][c] = acc1[t][v];
            atomicAdd(&s2[r][c], acc2[t][v] + bias[h * Cc + c]);
        }
    __syncthreads();

    // attention dot products (64 threads: one per (head,row))
    if (tid < 64) {
        int hd = tid >> 4, r = tid & 15;
        float ds = 0.0f, dn = 0.0f;
        for (int c = 0; c < Cc; ++c) {
            float v = fts[hd][r][c];
            ds += v * a_self [hd * Cc + c];
            dn += v * a_neigh[hd * Cc + c];
        }
        att_s[hd * Nn + i0 + r] = ds;
        att_n[hd * Nn + i0 + r] = dn;
    }

    // featsT (bf16, [h][c][n]) and S2 global writes
    for (int idx = tid; idx < Hh * Cc * 16; idx += 128) {
        int hd = idx >> 10, c = (idx >> 4) & 63, r = idx & 15;
        featsT[(size_t)(hd * Cc + c) * Nn + i0 + r] = bf16b(fts[hd][r][c]);
    }
    for (int idx = tid; idx < 16 * Cc; idx += 128) {
        int r = idx >> 6, c = idx & 63;
        S2[(size_t)(i0 + r) * Cc + c] = s2[r][c];
    }
}

// ---------------------------------------------------------------------------
// Kernel 2: fused masked softmax + attn@feats (flash style), one pass over A.
// Block = 128 threads; wave h = head h; block handles rows [i0, i0+16).
// Online softmax state per row is kept in lane pairs (L, L+16), combined by
// __shfl_xor(.,16). The accumulator rescale (8 shuffles + 32 multiplies) only
// runs when some row's max actually grew (ballot-guarded): with 1% adjacency
// the tile max is ~-1e10 almost always, so the guard skips nearly every time.
// ---------------------------------------------------------------------------
__global__ __launch_bounds__(128) void gat_attn(
    const float* __restrict__ A, const unsigned short* __restrict__ featsT,
    const float* __restrict__ att_s, const float* __restrict__ att_n,
    const float* __restrict__ S2, float* __restrict__ out)
{
    __shared__ float osum[16][Cc];

    const int tid  = threadIdx.x;
    const int h    = tid >> 5;
    const int lane = tid & 31;
    const int hlf  = lane >> 4;
    const int l16  = lane & 15;
    const int i0   = blockIdx.x * 16;
    const int kb0  = 8 * hlf;                    // this half-wave's k-run base

    for (int idx = tid; idx < 16 * Cc; idx += 128)
        ((float*)osum)[idx] = 0.0f;
    __syncthreads();

    const float  asv  = att_s[h * Nn + i0 + l16];
    const float* Arow = A + (size_t)(i0 + l16) * Nn;
    const float* anp  = att_n + h * Nn;

    float m = -3.0e38f, l = 0.0f;
    v8f acc[4];
    #pragma unroll
    for (int t = 0; t < 4; ++t)
        #pragma unroll
        for (int v = 0; v < 8; ++v) acc[t][v] = 0.0f;

    #pragma unroll 2
    for (int j0 = 0; j0 < Nn; j0 += 32) {
        __builtin_prefetch(Arow + j0 + 512, 0, 0);   // global_prefetch_b8

        // adjacency mask + neighbor logits for this lane's 16 k's:
        // runs [kb0,kb0+8) and [16+kb0,16+kb0+8)
        float4 a0 = *(const float4*)(Arow + j0 + kb0);
        float4 a1 = *(const float4*)(Arow + j0 + kb0 + 4);
        float4 a2 = *(const float4*)(Arow + j0 + 16 + kb0);
        float4 a3 = *(const float4*)(Arow + j0 + 16 + kb0 + 4);
        float4 n0 = *(const float4*)(anp + j0 + kb0);
        float4 n1 = *(const float4*)(anp + j0 + kb0 + 4);
        float4 n2 = *(const float4*)(anp + j0 + 16 + kb0);
        float4 n3 = *(const float4*)(anp + j0 + 16 + kb0 + 4);

        float av[16] = {a0.x,a0.y,a0.z,a0.w, a1.x,a1.y,a1.z,a1.w,
                        a2.x,a2.y,a2.z,a2.w, a3.x,a3.y,a3.z,a3.w};
        float nv[16] = {n0.x,n0.y,n0.z,n0.w, n1.x,n1.y,n1.z,n1.w,
                        n2.x,n2.y,n2.z,n2.w, n3.x,n3.y,n3.z,n3.w};

        float wq[16];
        float pm = -3.0e38f;
        #pragma unroll
        for (int q = 0; q < 16; ++q) {
            float z = asv + nv[q];
            z = z > 0.0f ? z : 0.2f * z;            // LeakyReLU(0.2)
            z += NEGC * (1.0f - av[q]);             // adjacency mask
            wq[q] = z;
            pm = fmaxf(pm, z);
        }
        float mn = fmaxf(pm, __shfl_xor(pm, 16));   // this tile's row max

        if (__ballot(mn > m)) {                     // wave-uniform, rare
            float mnew  = fmaxf(m, mn);
            float scale = __expf(m - mnew);
            l *= scale;
            float sv[8];
            #pragma unroll
            for (int v = 0; v < 8; ++v) sv[v] = __shfl(scale, v + 8 * hlf);
            #pragma unroll
            for (int t = 0; t < 4; ++t)
                #pragma unroll
                for (int v = 0; v < 8; ++v) acc[t][v] *= sv[v];
            m = mnew;
        }

        float ps = 0.0f;
        #pragma unroll
        for (int q = 0; q < 16; ++q) {
            float w = __expf(wq[q] - m);
            wq[q] = w;
            ps += w;
        }
        l += ps + __shfl_xor(ps, 16);

        // pack weights into A-fragment (bf16): element e covers k=frag_k(e,hlf)
        BFrag af;
        #pragma unroll
        for (int e = 0; e < 16; ++e) {
            int v = e >> 1;
            int q = (v >> 2) * 8 + (v & 3) * 2 + (e & 1);
            af.us[e] = bf16b(wq[q]);
        }

        // 4 WMMA against featsT (bf16) B-tiles; A-fragment reused
        #pragma unroll
        for (int t = 0; t < 4; ++t) {
            BFrag bf;
            const unsigned short* fp =
                featsT + (size_t)(h * Cc + t * 16 + l16) * Nn + j0;
            bf.q[0] = *(const uint4*)(fp + kb0);        // k = kb0 .. kb0+7
            bf.q[1] = *(const uint4*)(fp + 16 + kb0);   // k = 16+kb0 .. +7
            acc[t] = __builtin_amdgcn_wmma_f32_16x16x32_bf16(
                false, af.v, false, bf.v, (short)0, acc[t], false, false);
        }
    }

    // normalize by row sums and reduce across heads into LDS
    float linv[8];
    #pragma unroll
    for (int v = 0; v < 8; ++v)
        linv[v] = 1.0f / __shfl(l, v + 8 * hlf);
    #pragma unroll
    for (int t = 0; t < 4; ++t)
        #pragma unroll
        for (int v = 0; v < 8; ++v)
            atomicAdd(&osum[v + 8 * hlf][t * 16 + l16], acc[t][v] * linv[v]);
    __syncthreads();

    // out = relu(mean over heads + feats2-term)
    for (int idx = tid; idx < 16 * Cc; idx += 128) {
        int r = idx >> 6, c = idx & 63;
        float o = 0.25f * (osum[r][c] + S2[(size_t)(i0 + r) * Cc + c]);
        out[(size_t)(i0 + r) * Cc + c] = o > 0.0f ? o : 0.0f;
    }
}

// ---------------------------------------------------------------------------
extern "C" void kernel_launch(void* const* d_in, const int* in_sizes, int n_in,
                              void* d_out, int out_size, void* d_ws,
                              size_t ws_size, hipStream_t stream)
{
    const float* X       = (const float*)d_in[0];
    const float* A       = (const float*)d_in[1];
    const float* W1      = (const float*)d_in[2];
    const float* W2      = (const float*)d_in[3];
    const float* a_self  = (const float*)d_in[4];
    const float* a_neigh = (const float*)d_in[5];
    const float* bias    = (const float*)d_in[6];
    float* out = (float*)d_out;

    char* ws = (char*)d_ws;
    const size_t featsT_bytes = (size_t)Hh * Cc * Nn * 2;   // 3,145,728
    const size_t att_bytes    = (size_t)Hh * Nn * 4;        //    98,304
    const size_t S2_bytes     = (size_t)Nn * Cc * 4;        // 1,572,864
    unsigned short* featsT = (unsigned short*)ws;
    float* att_s = (float*)(ws + featsT_bytes);
    float* att_n = (float*)(ws + featsT_bytes + att_bytes);
    float* S2    = (float*)(ws + featsT_bytes + 2 * att_bytes);
    unsigned short* Wf =
        (unsigned short*)(ws + featsT_bytes + 2 * att_bytes + S2_bytes);

    gat_prep_w<<<256, 256, 0, stream>>>(W1, W2, Wf);
    gat_proj<<<Nn / 16, 128, 0, stream>>>(X, Wf, a_self, a_neigh, bias,
                                          featsT, att_s, att_n, S2);
    gat_attn<<<Nn / 16, 128, 0, stream>>>(A, featsT, att_s, att_n, S2, out);
}
